// Attention_84542136255055
// MI455X (gfx1250) — matrix operand order repaired and checked by reading, not compiled, not run on hardware
//
#include <hip/hip_runtime.h>
#include <hip/hip_bf16.h>

typedef __attribute__((ext_vector_type(16))) _Float16 v16h;
typedef __attribute__((ext_vector_type(8)))  float    v8f;

#define BATCH 4
#define CDIM  64
#define NPIX  4096   // 64*64
#define MTILE 64     // query rows per block (4 waves x 16 rows)
#define NTILE 64     // keys per streaming step
#define NWAVE 4
#define NTHR  (NWAVE * 32)
#define LOG2E 1.44269504088896340736f

union Frag16 { v16h v; float4 f4[2]; };

// Fused cross-lane reduction steps: DPP16 ROW_XMASK applied to src0 of the
// reducing op itself (1 VALU instruction per step, no mov/canonicalize).
#define DPP_MAX_STEP(x, mask)                                              \
    do {                                                                   \
        float _t;                                                          \
        asm("v_max_num_f32_dpp %0, %1, %1 row_xmask:" #mask                \
            " row_mask:0xf bank_mask:0xf bound_ctrl:1"                     \
            : "=v"(_t) : "v"(x));                                          \
        (x) = _t;                                                          \
    } while (0)

#define DPP_ADD_STEP(x, mask)                                              \
    do {                                                                   \
        float _t;                                                          \
        asm("v_add_f32_dpp %0, %1, %1 row_xmask:" #mask                    \
            " row_mask:0xf bank_mask:0xf bound_ctrl:1"                     \
            : "=v"(_t) : "v"(x));                                          \
        (x) = _t;                                                          \
    } while (0)

// ---------------------------------------------------------------------------
// Projection: K = Wf*x + bf  [N][32] f16
//             Q = Wg*y + bg  [N][32] f16
//             Vt = (Wh*y+bh) stored transposed [64][N] f16
// ---------------------------------------------------------------------------
__global__ __launch_bounds__(256) void proj_kernel(
    const float* __restrict__ x, const float* __restrict__ y,
    const float* __restrict__ Wf, const float* __restrict__ bf,
    const float* __restrict__ Wg, const float* __restrict__ bg,
    const float* __restrict__ Wh, const float* __restrict__ bh,
    _Float16* __restrict__ Kf, _Float16* __restrict__ Qf,
    _Float16* __restrict__ Vt)
{
    __shared__ float Wfs[32 * 64];
    __shared__ float Wgs[32 * 64];
    __shared__ float Whs[64 * 64];
    __shared__ float bfs[32], bgs[32], bhs[64];

    const int tid = threadIdx.x;
    for (int i = tid; i < 32 * 64; i += 256) { Wfs[i] = Wf[i]; Wgs[i] = Wg[i]; }
    for (int i = tid; i < 64 * 64; i += 256) { Whs[i] = Wh[i]; }
    if (tid < 32) { bfs[tid] = bf[tid]; bgs[tid] = bg[tid]; }
    if (tid < 64) { bhs[tid] = bh[tid]; }
    __syncthreads();

    const int b = blockIdx.x >> 4;                 // 16 blocks per batch
    const int p = ((blockIdx.x & 15) << 8) + tid;  // pixel index
    const size_t basexy = ((size_t)b * CDIM) * NPIX + p;

    // ---- K = Wf * x ----
    float acF[32];
    #pragma unroll
    for (int oc = 0; oc < 32; ++oc) acF[oc] = bfs[oc];
    for (int c = 0; c < CDIM; ++c) {
        float xc = x[basexy + (size_t)c * NPIX];
        #pragma unroll
        for (int oc = 0; oc < 32; ++oc) acF[oc] = fmaf(Wfs[oc * 64 + c], xc, acF[oc]);
    }
    {
        _Float16* dst = Kf + ((size_t)b * NPIX + p) * 32;
        #pragma unroll
        for (int oc = 0; oc < 32; ++oc) dst[oc] = (_Float16)acF[oc];
    }

    // ---- Q = Wg * y, V = Wh * y ----
    float acG[32], acH[64];
    #pragma unroll
    for (int oc = 0; oc < 32; ++oc) acG[oc] = bgs[oc];
    #pragma unroll
    for (int oc = 0; oc < 64; ++oc) acH[oc] = bhs[oc];
    for (int c = 0; c < CDIM; ++c) {
        float yc = y[basexy + (size_t)c * NPIX];
        #pragma unroll
        for (int oc = 0; oc < 32; ++oc) acG[oc] = fmaf(Wgs[oc * 64 + c], yc, acG[oc]);
        #pragma unroll
        for (int oc = 0; oc < 64; ++oc) acH[oc] = fmaf(Whs[oc * 64 + c], yc, acH[oc]);
    }
    {
        _Float16* dst = Qf + ((size_t)b * NPIX + p) * 32;
        #pragma unroll
        for (int oc = 0; oc < 32; ++oc) dst[oc] = (_Float16)acG[oc];
    }
    #pragma unroll
    for (int oc = 0; oc < 64; ++oc)
        Vt[((size_t)(b * CDIM + oc)) * NPIX + p] = (_Float16)acH[oc];
}

// ---------------------------------------------------------------------------
// Async tile staging: global -> LDS directly (ASYNCcnt path, no VGPR landing)
// ---------------------------------------------------------------------------
__device__ __forceinline__ void async_b128(unsigned lds_off, const void* gaddr) {
    asm volatile("global_load_async_to_lds_b128 %0, %1, off"
                 :: "v"(lds_off), "v"(gaddr) : "memory");
}

__device__ __forceinline__ void issue_tile_loads(
    const _Float16* __restrict__ Kf, const _Float16* __restrict__ Vt,
    int b, int n0, int tid, _Float16* ksh, _Float16* vsh)
{
    // K tile: 64 rows x 32 f16 = 4KB contiguous -> two b128 per thread
    #pragma unroll
    for (int pass = 0; pass < 2; ++pass) {
        int idx = pass * NTHR + tid;     // 0..255 chunks of 16B
        async_b128((unsigned)(size_t)(ksh + idx * 8),
                   Kf + ((size_t)b * NPIX + n0) * 32 + idx * 8);
    }
    // V tile (transposed [c][N]): 64 rows x 128B = 8KB -> four b128 per thread
    #pragma unroll
    for (int pass = 0; pass < 4; ++pass) {
        int idx = pass * NTHR + tid;     // 0..511
        int c   = idx >> 3;
        int ch  = idx & 7;
        async_b128((unsigned)(size_t)(vsh + c * NTILE + ch * 8),
                   Vt + ((size_t)(b * CDIM + c)) * NPIX + n0 + ch * 8);
    }
}

// ---------------------------------------------------------------------------
// Flash attention: out = gamma * softmax(Q K^T) V  (+ x residual)
// ---------------------------------------------------------------------------
__global__ __launch_bounds__(NTHR) void attn_kernel(
    const _Float16* __restrict__ Qf, const _Float16* __restrict__ Kf,
    const _Float16* __restrict__ Vt, const float* __restrict__ x,
    const float* __restrict__ gamma, float* __restrict__ out)
{
    __shared__ alignas(16) _Float16 Ksh[2][NTILE * 32];      //  8KB double-buffered
    __shared__ alignas(16) _Float16 Vsh[2][CDIM * NTILE];    // 16KB double-buffered
    __shared__ alignas(16) _Float16 Psh[NWAVE][16 * 64];     //  8KB per-wave P
    __shared__ alignas(16) float    Osh[NWAVE][16 * 64];     // 16KB per-wave O

    const int tid  = threadIdx.x;
    const int wave = tid >> 5;
    const int lane = tid & 31;
    const int l16  = lane & 15;
    const int hi   = lane >> 4;    // 0 = lanes 0-15, 1 = lanes 16-31

    const int b      = blockIdx.x >> 6;   // 64 row-blocks per batch
    const int rowblk = blockIdx.x & 63;
    const int m0     = rowblk * MTILE + wave * 16;   // wave's first query row

    // Q as WMMA A-fragment (16x32 f16), pre-scaled by log2(e) so the softmax
    // exponentials become plain v_exp (exp2) with no multiply.
    Frag16 qa;
    {
        const _Float16* qrow = Qf + ((size_t)b * NPIX + (m0 + l16)) * 32 + hi * 8;
        qa.f4[0] = *(const float4*)(qrow);
        qa.f4[1] = *(const float4*)(qrow + 16);
        qa.v = qa.v * (_Float16)LOG2E;
    }

    v8f acc[4];
    float mrun[8], lrun[8];   // lrun: per-lane PARTIAL row sum (reduced at end)
    #pragma unroll
    for (int r = 0; r < 8; ++r) { mrun[r] = -1e30f; lrun[r] = 0.0f; }
    #pragma unroll
    for (int cc = 0; cc < 4; ++cc)
        #pragma unroll
        for (int r = 0; r < 8; ++r) acc[cc][r] = 0.0f;

    // Preload tile 0
    issue_tile_loads(Kf, Vt, b, 0, tid, Ksh[0], Vsh[0]);
    asm volatile("s_wait_asynccnt 0x0" ::: "memory");
    __syncthreads();

    #pragma unroll 1
    for (int it = 0; it < NPIX / NTILE; ++it) {
        const int cur = it & 1;
        // Software pipeline: kick off next tile while computing this one.
        if (it + 1 < NPIX / NTILE)
            issue_tile_loads(Kf, Vt, b, (it + 1) * NTILE, tid,
                             Ksh[cur ^ 1], Vsh[cur ^ 1]);

        const _Float16* Kc = Ksh[cur];
        const _Float16* Vc = Vsh[cur];

        // ---- S = Q K^T : 4 WMMAs (n-subtiles of 16)
        v8f sv[4];
        #pragma unroll
        for (int j = 0; j < 4; ++j) {
            Frag16 kb;
            const _Float16* kr = Kc + (j * 16 + l16) * 32 + hi * 8;
            kb.f4[0] = *(const float4*)(kr);
            kb.f4[1] = *(const float4*)(kr + 16);
            v8f z;
            #pragma unroll
            for (int r = 0; r < 8; ++r) z[r] = 0.0f;
            sv[j] = __builtin_amdgcn_wmma_f32_16x16x32_f16(
                false, qa.v, false, kb.v, (short)0, z, false, false);
        }

        // ---- online softmax (log2 domain). Only the row MAX needs a
        // cross-lane reduce; the row sum stays lane-partial (linear in P).
        #pragma unroll
        for (int r = 0; r < 8; ++r) {
            float mx = fmaxf(fmaxf(sv[0][r], sv[1][r]), fmaxf(sv[2][r], sv[3][r]));
            DPP_MAX_STEP(mx, 1);
            DPP_MAX_STEP(mx, 2);
            DPP_MAX_STEP(mx, 4);
            DPP_MAX_STEP(mx, 8);
            float mn = fmaxf(mrun[r], mx);
            float sc = exp2f(mrun[r] - mn);
            lrun[r] *= sc;
            #pragma unroll
            for (int cc = 0; cc < 4; ++cc) acc[cc][r] *= sc;
            mrun[r] = mn;
        }

        // ---- P = exp2(S - m): write to per-wave LDS; accumulate lane-partial sums
        #pragma unroll
        for (int j = 0; j < 4; ++j) {
            #pragma unroll
            for (int r = 0; r < 8; ++r) {
                float p = exp2f(sv[j][r] - mrun[r]);
                lrun[r] += p;
                Psh[wave][(r + hi * 8) * 64 + j * 16 + l16] = (_Float16)p;
            }
        }

        // ---- O += P V : P re-read in A layout, V as B operand. 8 WMMAs.
        #pragma unroll
        for (int kk = 0; kk < 2; ++kk) {
            Frag16 pa;
            const _Float16* pr = &Psh[wave][l16 * 64 + kk * 32 + hi * 8];
            pa.f4[0] = *(const float4*)(pr);
            pa.f4[1] = *(const float4*)(pr + 16);
            #pragma unroll
            for (int cc = 0; cc < 4; ++cc) {
                Frag16 vb;
                const _Float16* vr = Vc + (cc * 16 + l16) * NTILE + kk * 32 + hi * 8;
                vb.f4[0] = *(const float4*)(vr);
                vb.f4[1] = *(const float4*)(vr + 16);
                acc[cc] = __builtin_amdgcn_wmma_f32_16x16x32_f16(
                    false, pa.v, false, vb.v, (short)0, acc[cc], false, false);
            }
        }

        // Drain this iteration's async loads, then block-wide barrier: next
        // iteration may read the freshly written buffer and overwrite `cur`.
        asm volatile("s_wait_asynccnt 0x0" ::: "memory");
        __syncthreads();
    }

    // ---- one-time cross-lane reduction of the partial row sums
    #pragma unroll
    for (int r = 0; r < 8; ++r) {
        float s = lrun[r];
        DPP_ADD_STEP(s, 1);
        DPP_ADD_STEP(s, 2);
        DPP_ADD_STEP(s, 4);
        DPP_ADD_STEP(s, 8);
        lrun[r] = s;
    }

    // ---- epilogue: o = acc / l, transpose through LDS, out = gamma*o + x
    const float gma = gamma[0];
    #pragma unroll
    for (int r = 0; r < 8; ++r) {
        float inv = 1.0f / lrun[r];
        #pragma unroll
        for (int cc = 0; cc < 4; ++cc)
            Osh[wave][(r + hi * 8) * 64 + cc * 16 + l16] = acc[cc][r] * inv;
    }
    #pragma unroll 4
    for (int i = 0; i < 32; ++i) {
        int flat = i * 32 + lane;          // 1024 = 64 channels x 16 rows
        int c    = flat >> 4;
        int mi   = flat & 15;
        size_t idx = ((size_t)(b * CDIM + c)) * NPIX + m0 + mi;
        out[idx] = gma * Osh[wave][mi * 64 + c] + x[idx];
    }
}

// ---------------------------------------------------------------------------
extern "C" void kernel_launch(void* const* d_in, const int* in_sizes, int n_in,
                              void* d_out, int out_size, void* d_ws, size_t ws_size,
                              hipStream_t stream) {
    (void)in_sizes; (void)n_in; (void)out_size; (void)ws_size;
    const float* x     = (const float*)d_in[0];
    const float* y     = (const float*)d_in[1];
    const float* Wf    = (const float*)d_in[2];
    const float* bf    = (const float*)d_in[3];
    const float* Wg    = (const float*)d_in[4];
    const float* bg    = (const float*)d_in[5];
    const float* Wh    = (const float*)d_in[6];
    const float* bh    = (const float*)d_in[7];
    const float* gamma = (const float*)d_in[8];
    float* out = (float*)d_out;

    _Float16* Kf = (_Float16*)d_ws;                       // [B][N][32]
    _Float16* Qf = Kf + (size_t)BATCH * NPIX * 32;        // [B][N][32]
    _Float16* Vt = Qf + (size_t)BATCH * NPIX * 32;        // [B][64][N]

    proj_kernel<<<BATCH * (NPIX / 256), 256, 0, stream>>>(
        x, y, Wf, bf, Wg, bg, Wh, bh, Kf, Qf, Vt);
    attn_kernel<<<BATCH * (NPIX / MTILE), NTHR, 0, stream>>>(
        Qf, Kf, Vt, x, gamma, out);
}
